// SoftmaxMoELoRALinear_44513041055939
// MI455X (gfx1250) — matrix operand order repaired and checked by reading, not compiled, not run on hardware
//
#include <hip/hip_runtime.h>
#include <hip/hip_bf16.h>
#include <math.h>

typedef __bf16 bf16_t;
typedef __bf16 v8bf  __attribute__((ext_vector_type(8)));
typedef __bf16 v16bf __attribute__((ext_vector_type(16)));
typedef float  v8f   __attribute__((ext_vector_type(8)));

#define D_DIM 2048
#define O_DIM 2048
#define M_TOT 8192
#define E_NUM 8
#define R_NUM 16
#define ER    128
#define SCALING_F 1.0f   // 16.0 / 16.0

#define BM 128
#define BN 128
#define BK 32
#define LDSTR 40                 // bf16 per LDS row: 32 data + 8 pad (80B, 16B-aligned)
#define ABUF_E (BM * LDSTR)      // 5120 bf16 elements per A buffer
#define BBUF_E (BN * LDSTR)      // 5120 bf16 elements per B buffer

// ---------------------------------------------------------------------------
// Precision conversion / layout kernels
// ---------------------------------------------------------------------------
__global__ void cvt_f32_bf16(const float* __restrict__ in, bf16_t* __restrict__ out, int n) {
  for (int i = blockIdx.x * blockDim.x + threadIdx.x; i < n; i += gridDim.x * blockDim.x)
    out[i] = (bf16_t)in[i];
}

// aT[er][d] = A[e][d][r]  -> shape (128, 2048), K(d)-contiguous per column er
__global__ void build_aT(const float* __restrict__ A, bf16_t* __restrict__ aT) {
  const int n = ER * D_DIM;
  for (int i = blockIdx.x * blockDim.x + threadIdx.x; i < n; i += gridDim.x * blockDim.x) {
    int er = i / D_DIM, d = i % D_DIM;
    int e = er >> 4, r = er & 15;
    aT[i] = (bf16_t)A[((size_t)(e * D_DIM + d)) * R_NUM + r];
  }
}

// bT[o][er] = Bexp[e][r][o] -> shape (2048, 128), K(er)-contiguous per column o
__global__ void build_bT(const float* __restrict__ Bexp, bf16_t* __restrict__ bT) {
  const int n = O_DIM * ER;
  for (int i = blockIdx.x * blockDim.x + threadIdx.x; i < n; i += gridDim.x * blockDim.x) {
    int o = i / ER, er = i % ER;
    int e = er >> 4, r = er & 15;
    bT[i] = (bf16_t)Bexp[((size_t)(e * R_NUM + r)) * O_DIM + o];
  }
}

// ---------------------------------------------------------------------------
// Router: full f32 logits + top-2 masked softmax (one wave32 per token row)
// ---------------------------------------------------------------------------
__global__ void __launch_bounds__(256)
router_kernel(const float* __restrict__ x, const float* __restrict__ Wr,
              float* __restrict__ wgt) {
  const int wave = threadIdx.x >> 5;
  const int lane = threadIdx.x & 31;
  const int row  = blockIdx.x * 8 + wave;
  const float* xr = x + (size_t)row * D_DIM;

  float acc[E_NUM];
#pragma unroll
  for (int e = 0; e < E_NUM; ++e) acc[e] = 0.f;

  for (int d = lane; d < D_DIM; d += 32) {
    float xv = xr[d];
#pragma unroll
    for (int e = 0; e < E_NUM; ++e) acc[e] += xv * Wr[e * D_DIM + d];
  }
#pragma unroll
  for (int e = 0; e < E_NUM; ++e)
    for (int off = 16; off > 0; off >>= 1)
      acc[e] += __shfl_xor(acc[e], off, 32);

  if (lane == 0) {
    int i1 = 0; float m1 = acc[0];
#pragma unroll
    for (int e = 1; e < E_NUM; ++e) if (acc[e] > m1) { m1 = acc[e]; i1 = e; }
    int i2 = -1; float m2 = -3.4e38f;
#pragma unroll
    for (int e = 0; e < E_NUM; ++e) if (e != i1 && acc[e] > m2) { m2 = acc[e]; i2 = e; }
    float e2 = __expf(m2 - m1);
    float Z  = 1.f + e2;
#pragma unroll
    for (int e = 0; e < E_NUM; ++e)
      wgt[row * E_NUM + e] = (e == i1) ? (1.f / Z) : ((e == i2) ? (e2 / Z) : 0.f);
  }
}

// ---------------------------------------------------------------------------
// Async staging + WMMA building blocks
// ---------------------------------------------------------------------------
// LDS byte offset of a shared-memory pointer (AS3 pointers are 32-bit offsets)
__device__ __forceinline__ unsigned lds_off_of(void* p) {
  return (unsigned)(unsigned long long)(__attribute__((address_space(3))) void*)p;
}

__device__ __forceinline__ void wait_async0() {
#if __has_builtin(__builtin_amdgcn_s_wait_asynccnt)
  __builtin_amdgcn_s_wait_asynccnt(0);
#else
  asm volatile("s_wait_asynccnt 0x0" ::: "memory");
#endif
}

// Issue async global->LDS copies of a 128-row x 32-col bf16 tile.
// Each of the 256 threads moves 2 chunks of 16B (ASYNCcnt-tracked, no VGPR data).
__device__ __forceinline__ void issue_stage(const bf16_t* __restrict__ g, int pitch,
                                            int rowbase, int k0, unsigned s_bytes, int tid) {
#pragma unroll
  for (int i = 0; i < 2; ++i) {
    int c = tid + i * 256;      // 512 chunks of 8 bf16 (16B)
    int r = c >> 2, q = c & 3;
    const bf16_t* src = g + (size_t)(rowbase + r) * pitch + k0 + q * 8;
    unsigned dst = s_bytes + (unsigned)((r * LDSTR + q * 8) * 2);
    asm volatile("global_load_async_to_lds_b128 %0, %1, off"
                 :: "v"(dst), "v"(src)
                 : "memory");
  }
}

// One BK=32 step: each wave does 4x2 tiles of v_wmma_f32_16x16x32_bf16.
__device__ __forceinline__ void wmma_step(const bf16_t* sA, const bf16_t* sB,
                                          int wm, int wn, int lane, v8f (&acc)[4][2]) {
  const int lh = lane >> 4;     // half-wave select
  const int li = lane & 15;
  v16bf a[4], b[2];
#pragma unroll
  for (int mt = 0; mt < 4; ++mt) {
    // A 16x32 bf16 fragment: lane = M row; lanes 0-15 K 0..7 & 16..23, lanes 16-31 K 8..15 & 24..31
    const bf16_t* p = sA + (wm * 64 + mt * 16 + li) * LDSTR + lh * 8;
    v8bf lo = *(const v8bf*)p;
    v8bf hi = *(const v8bf*)(p + 16);
    a[mt] = __builtin_shufflevector(lo, hi, 0, 1, 2, 3, 4, 5, 6, 7,
                                            8, 9, 10, 11, 12, 13, 14, 15);
  }
#pragma unroll
  for (int nt = 0; nt < 2; ++nt) {
    // B 32x16 bf16 fragment: lane = N col; lanes 0-15 K 0..7 & 16..23, lanes 16-31 K 8..15 & 24..31
    const bf16_t* p = sB + (wn * 32 + nt * 16 + li) * LDSTR + lh * 8;
    v8bf lo = *(const v8bf*)p;
    v8bf hi = *(const v8bf*)(p + 16);
    b[nt] = __builtin_shufflevector(lo, hi, 0, 1, 2, 3, 4, 5, 6, 7,
                                            8, 9, 10, 11, 12, 13, 14, 15);
  }
#pragma unroll
  for (int mt = 0; mt < 4; ++mt)
#pragma unroll
    for (int nt = 0; nt < 2; ++nt)
      acc[mt][nt] = __builtin_amdgcn_wmma_f32_16x16x32_bf16(
          false, a[mt], false, b[nt], (short)0, acc[mt][nt], false, false);
}

// ---------------------------------------------------------------------------
// GEMM1: U = (x @ Acat) * router_weight * SCALING  -> bf16 (8192 x 128)
// Double-buffered async-LDS pipeline.
// ---------------------------------------------------------------------------
__global__ void __launch_bounds__(256)
lora_down_gemm(const bf16_t* __restrict__ xh, const bf16_t* __restrict__ aT,
               const float* __restrict__ wgt, bf16_t* __restrict__ uh) {
  __shared__ bf16_t smem[2 * ABUF_E + 2 * BBUF_E];

  const int tid = threadIdx.x, lane = tid & 31, wave = tid >> 5;
  const int wm = wave >> 2, wn = wave & 3;
  const int rowBase = blockIdx.x * BM;

  const unsigned base = lds_off_of(smem);
  const unsigned aOff[2] = {base, base + ABUF_E * 2u};
  const unsigned bOff[2] = {base + 4u * ABUF_E, base + 4u * ABUF_E + BBUF_E * 2u};

  v8f acc[4][2];
#pragma unroll
  for (int mt = 0; mt < 4; ++mt)
#pragma unroll
    for (int nt = 0; nt < 2; ++nt) acc[mt][nt] = {};

  const int NSTEP = D_DIM / BK;   // 64
  issue_stage(xh, D_DIM, rowBase, 0, aOff[0], tid);
  issue_stage(aT, D_DIM, 0,       0, bOff[0], tid);
  wait_async0();
  __syncthreads();

  for (int ks = 0; ks < NSTEP; ++ks) {
    int cur = ks & 1, nxt = cur ^ 1;
    if (ks + 1 < NSTEP) {
      int k0 = (ks + 1) * BK;
      issue_stage(xh, D_DIM, rowBase, k0, aOff[nxt], tid);
      issue_stage(aT, D_DIM, 0,       k0, bOff[nxt], tid);
    }
    wmma_step(smem + cur * ABUF_E, smem + 2 * ABUF_E + cur * BBUF_E, wm, wn, lane, acc);
    wait_async0();
    __syncthreads();
  }

  const int lh = lane >> 4, li = lane & 15;
#pragma unroll
  for (int mt = 0; mt < 4; ++mt) {
#pragma unroll
    for (int nt = 0; nt < 2; ++nt) {
      int er = wn * 32 + nt * 16 + li;
      int e  = er >> 4;
      int r0 = rowBase + wm * 64 + mt * 16 + lh * 8;
#pragma unroll
      for (int v = 0; v < 8; ++v) {
        int row = r0 + v;
        float wv = SCALING_F * wgt[row * E_NUM + e];
        uh[(size_t)row * ER + er] = (bf16_t)(acc[mt][nt][v] * wv);
      }
    }
  }
}

// ---------------------------------------------------------------------------
// GEMM2: out = x @ W^T + U @ Bcat + bias  -> f32 (8192 x 2048)
// K = 2048 (base weight) then K = 128 (LoRA up), same accumulators.
// ---------------------------------------------------------------------------
#define NSTEP1 (D_DIM / BK)      // 64
#define NSTEP2 (ER / BK)         // 4
#define NSTEPT (NSTEP1 + NSTEP2) // 68

__device__ __forceinline__ void issue_step_fused(int ks,
    const bf16_t* __restrict__ xh, const bf16_t* __restrict__ wh,
    const bf16_t* __restrict__ uh, const bf16_t* __restrict__ bT,
    int rowBase, int colBase, unsigned aOff, unsigned bOff, int tid) {
  if (ks < NSTEP1) {
    int k0 = ks * BK;
    issue_stage(xh, D_DIM, rowBase, k0, aOff, tid);
    issue_stage(wh, D_DIM, colBase, k0, bOff, tid);   // W is (O,D): row o = B column o
  } else {
    int k0 = (ks - NSTEP1) * BK;
    issue_stage(uh, ER, rowBase, k0, aOff, tid);
    issue_stage(bT, ER, colBase, k0, bOff, tid);
  }
}

__global__ void __launch_bounds__(256)
fused_out_gemm(const bf16_t* __restrict__ xh, const bf16_t* __restrict__ wh,
               const bf16_t* __restrict__ uh, const bf16_t* __restrict__ bT,
               const float* __restrict__ bias, float* __restrict__ out) {
  __shared__ bf16_t smem[2 * ABUF_E + 2 * BBUF_E];

  const int tid = threadIdx.x, lane = tid & 31, wave = tid >> 5;
  const int wm = wave >> 2, wn = wave & 3;
  const int rowBase = blockIdx.x * BM;
  const int colBase = blockIdx.y * BN;

  const unsigned base = lds_off_of(smem);
  const unsigned aOff[2] = {base, base + ABUF_E * 2u};
  const unsigned bOff[2] = {base + 4u * ABUF_E, base + 4u * ABUF_E + BBUF_E * 2u};

  v8f acc[4][2];
#pragma unroll
  for (int mt = 0; mt < 4; ++mt)
#pragma unroll
    for (int nt = 0; nt < 2; ++nt) acc[mt][nt] = {};

  issue_step_fused(0, xh, wh, uh, bT, rowBase, colBase, aOff[0], bOff[0], tid);
  wait_async0();
  __syncthreads();

  for (int ks = 0; ks < NSTEPT; ++ks) {
    int cur = ks & 1, nxt = cur ^ 1;
    if (ks + 1 < NSTEPT)
      issue_step_fused(ks + 1, xh, wh, uh, bT, rowBase, colBase, aOff[nxt], bOff[nxt], tid);
    wmma_step(smem + cur * ABUF_E, smem + 2 * ABUF_E + cur * BBUF_E, wm, wn, lane, acc);
    wait_async0();
    __syncthreads();
  }

  const int lh = lane >> 4, li = lane & 15;
#pragma unroll
  for (int mt = 0; mt < 4; ++mt) {
#pragma unroll
    for (int nt = 0; nt < 2; ++nt) {
      int col = colBase + wn * 32 + nt * 16 + li;
      float bv = bias[col];
      int r0 = rowBase + wm * 64 + mt * 16 + lh * 8;
#pragma unroll
      for (int v = 0; v < 8; ++v)
        out[(size_t)(r0 + v) * O_DIM + col] = acc[mt][nt][v] + bv;
    }
  }
}

// ---------------------------------------------------------------------------
// Host-side launch
// ---------------------------------------------------------------------------
extern "C" void kernel_launch(void* const* d_in, const int* in_sizes, int n_in,
                              void* d_out, int out_size, void* d_ws, size_t ws_size,
                              hipStream_t stream) {
  (void)in_sizes; (void)n_in; (void)out_size; (void)ws_size;

  const float* x    = (const float*)d_in[0];  // (4,2048,2048)
  const float* W    = (const float*)d_in[1];  // (2048,2048)
  const float* bias = (const float*)d_in[2];  // (2048,)
  const float* Wr   = (const float*)d_in[3];  // (8,2048)
  const float* A    = (const float*)d_in[4];  // (8,2048,16)
  const float* Bexp = (const float*)d_in[5];  // (8,16,2048)
  float* out = (float*)d_out;

  char* ws = (char*)d_ws;
  float*  wgt = (float*)(ws);                              // 8192*8*4       = 262144 B
  bf16_t* xh  = (bf16_t*)(ws + 262144);                    // 8192*2048*2    = 33554432 B
  bf16_t* wh  = (bf16_t*)(ws + 262144 + 33554432);         // 2048*2048*2    = 8388608 B
  bf16_t* uh  = (bf16_t*)(ws + 42205184);                  // 8192*128*2     = 2097152 B
  bf16_t* aT  = (bf16_t*)(ws + 44302336);                  // 128*2048*2     = 524288 B
  bf16_t* bT  = (bf16_t*)(ws + 44826624);                  // 2048*128*2     = 524288 B

  cvt_f32_bf16<<<4096, 256, 0, stream>>>(x, xh, M_TOT * D_DIM);
  cvt_f32_bf16<<<4096, 256, 0, stream>>>(W, wh, O_DIM * D_DIM);
  build_aT<<<512, 256, 0, stream>>>(A, aT);
  build_bT<<<512, 256, 0, stream>>>(Bexp, bT);
  router_kernel<<<M_TOT / 8, 256, 0, stream>>>(x, Wr, wgt);
  lora_down_gemm<<<dim3(M_TOT / BM), 256, 0, stream>>>(xh, aT, wgt, uh);
  fused_out_gemm<<<dim3(M_TOT / BM, O_DIM / BN), 256, 0, stream>>>(xh, wh, uh, bT, bias, out);
}
